// MultiHeadSelfAttention_12962211299801
// MI455X (gfx1250) — matrix operand order repaired and checked by reading, not compiled, not run on hardware
//
#include <hip/hip_runtime.h>
#include <cstdint>

// ---------------------------------------------------------------------------
// Fused single-head attention block (B=4, S=2048, D=1024) for MI455X/gfx1250.
// Compute-bound (~137 GFLOP vs ~100MB HBM) -> all GEMMs on bf16 WMMA.
// Tile staging via Tensor Data Mover, double-buffered so DMA overlaps WMMA:
//   barrier -> issue TDM(next tile -> idle buf) -> s_wait_tensorcnt(1)
//           -> barrier -> WMMA on current buf.
// ---------------------------------------------------------------------------

#define EMBED 1024
#define SEQ   2048
#define NB    4
#define MTOT  (NB * SEQ)          // 8192 rows total

typedef __attribute__((ext_vector_type(16))) __bf16 bf16x16;
typedef __attribute__((ext_vector_type(8)))  float  f32x8;

union Frag16 { bf16x16 v; unsigned int u[8]; };

#if defined(__has_builtin)
#if __has_builtin(__builtin_amdgcn_tensor_load_to_lds) && \
    __has_builtin(__builtin_amdgcn_s_wait_tensorcnt)
#define HAVE_TDM 1
#endif
#endif
#ifndef HAVE_TDM
#define HAVE_TDM 0
#endif

__device__ __forceinline__ unsigned short f2bf(float f) {
    union { float f; unsigned int u; } x; x.f = f;
    unsigned int r = x.u + 0x7fffu + ((x.u >> 16) & 1u);   // round-to-nearest-even
    return (unsigned short)(r >> 16);
}
__device__ __forceinline__ unsigned int f2bf2(float lo, float hi) {
    return (unsigned int)f2bf(lo) | ((unsigned int)f2bf(hi) << 16);
}

// A fragment (16x32 bf16): lane holds row=lane%16; 16 contiguous K at 2*hiA(0/16).
__device__ __forceinline__ void load_a_frag(Frag16& a, const unsigned short* rowp, int hiA) {
#pragma unroll
    for (int j = 0; j < 8; ++j) a.u[j] = *(const unsigned int*)(rowp + 2 * hiA + 2 * j);
}
// B fragment (32x16 bf16): lane holds col=lane%16; K pairs at hiB(0/16). Caller
// passes pointer to this lane's 32 contiguous K values (+hiB).
__device__ __forceinline__ void load_b_frag(Frag16& b, const unsigned short* colp) {
#pragma unroll
    for (int j = 0; j < 8; ++j) b.u[j] = *(const unsigned int*)(colp + 2 * j);
}

__device__ __forceinline__ f32x8 wmma_bf16(const Frag16& a, const Frag16& b, f32x8 c) {
    return __builtin_amdgcn_wmma_f32_16x16x32_bf16(false, a.v, false, b.v, (short)0, c,
                                                   false, false);
}

// ---------------------------------------------------------------------------
// TDM: load a 2-D tile of 2-byte elements (tile_rows x tile_cols, row stride in
// elements) from global into LDS, row-major packed. D# per ISA 08_async_tensor
// §8. Issue from ONE wave only (uniform branch gates it; EXEC ignored by TDM).
// ---------------------------------------------------------------------------
#if HAVE_TDM
typedef __attribute__((ext_vector_type(4))) unsigned int u32x4;
typedef __attribute__((ext_vector_type(8))) int i32x8;
typedef __attribute__((ext_vector_type(4))) int i32x4;

__device__ __forceinline__ void tdm_load_tile_b16(
    const unsigned short* g, unsigned short* lds,
    unsigned tile_cols, unsigned tile_rows, unsigned row_stride /* elements */)
{
    unsigned long long ga = (unsigned long long)(uintptr_t)g;
    unsigned int la = (unsigned int)(uintptr_t)lds;   // LDS aperture: addr[31:0]
    u32x4 g0;
    g0[0] = 1u;                                           // count=1, user descriptor
    g0[1] = la;                                           // lds_addr
    g0[2] = (unsigned int)ga;                             // global_addr[31:0]
    g0[3] = (unsigned int)((ga >> 32) & 0x1ffffffu) | (2u << 30);  // ga[56:32]|type=2
    i32x8 g1;
    g1[0] = 1 << 16;                                      // data_size=1 (2 bytes)
    g1[1] = (int)(tile_cols << 16);                       // tensor_dim0[15:0] << 16
    g1[2] = (int)((tile_cols >> 16) | (tile_rows << 16)); // dim0 hi | tensor_dim1 lo
    g1[3] = (int)((tile_rows >> 16) | (tile_cols << 16)); // dim1 hi | tile_dim0
    g1[4] = (int)tile_rows;                               // tile_dim1 (tile_dim2=0)
    g1[5] = (int)row_stride;                              // tensor_dim0_stride[31:0]
    g1[6] = (int)((row_stride & 0xffffu) << 16);          // d0str hi16=0 | d1str lo16
    g1[7] = (int)(row_stride >> 16);                      // d1str hi
    i32x4 z4 = {0, 0, 0, 0};
    i32x8 z8 = {0, 0, 0, 0, 0, 0, 0, 0};
    __builtin_amdgcn_tensor_load_to_lds(g0, g1, z4, z4, z8, 0);
}
// Wait with exactly one transfer left in flight (pipelined) or none (drain).
__device__ __forceinline__ void tdm_wait_pipe() { __builtin_amdgcn_s_wait_tensorcnt(1); }
__device__ __forceinline__ void tdm_wait_all()  { __builtin_amdgcn_s_wait_tensorcnt(0); }
#endif

// Fallback: plain dword copy of a bf16 tile into LDS, whole block cooperates.
__device__ __forceinline__ void copy_tile_b16(unsigned short* dst, const unsigned short* src,
                                              int rows, int row_dwords, int stride_elems,
                                              int tid) {
    unsigned int* d = (unsigned int*)dst;
    int total = rows * row_dwords;
    for (int i = tid; i < total; i += 256) {
        int r = i / row_dwords, c = i - r * row_dwords;
        d[i] = ((const unsigned int*)(src + (size_t)r * stride_elems))[c];
    }
}

// ---------------------------------------------------------------------------
// Kernel 0a: X f32 -> bf16 (one-time; keeps conversions out of GEMM loops)
// ---------------------------------------------------------------------------
__global__ __launch_bounds__(256) void cvt_bf16_kernel(const float2* __restrict__ src,
                                                       unsigned int* __restrict__ dst, int n) {
    int i = blockIdx.x * 256 + threadIdx.x;
    if (i < n) { float2 v = src[i]; dst[i] = f2bf2(v.x, v.y); }
}

// ---------------------------------------------------------------------------
// Kernel 0b: W[k][n] f32 -> Wt[n][k] bf16 for all 4 weights (one-time).
// Pre-transposed so GEMM B-tiles are contiguous-K rectangles (TDM friendly).
// ---------------------------------------------------------------------------
__global__ __launch_bounds__(256) void wtrans_kernel(
    const float* __restrict__ Wq, const float* __restrict__ Wk,
    const float* __restrict__ Wv, const float* __restrict__ Wo,
    unsigned short* __restrict__ Wt)
{
    __shared__ unsigned short T[64][66];
    const int z = blockIdx.z;
    const float* W = (z == 0) ? Wq : (z == 1) ? Wk : (z == 2) ? Wv : Wo;
    unsigned short* Out = Wt + (size_t)z * EMBED * EMBED;
    const int k0 = blockIdx.x * 64, n0 = blockIdx.y * 64;
    const int tid = threadIdx.x;
#pragma unroll
    for (int j = 0; j < 16; ++j) {
        int i = j * 256 + tid;
        int k = i >> 6, n = i & 63;
        T[k][n] = f2bf(W[(size_t)(k0 + k) * EMBED + n0 + n]);
    }
    __syncthreads();
#pragma unroll
    for (int j = 0; j < 8; ++j) {
        int i = j * 256 + tid;            // 2048 output dwords
        int n = i >> 5, k2 = i & 31;
        unsigned int v = (unsigned int)T[k2 * 2][n] | ((unsigned int)T[k2 * 2 + 1][n] << 16);
        ((unsigned int*)(Out + (size_t)(n0 + n) * EMBED + k0))[k2] = v;
    }
}

// ---------------------------------------------------------------------------
// Kernel 1: Q/K/V = Xbf * Wt^T + b (bf16 in/out). grid (16, 64, 3).
// Double-buffered TDM pipeline: wave0 streams A tiles, wave1 streams B tiles.
// ---------------------------------------------------------------------------
__global__ __launch_bounds__(256) void qkv_gemm_kernel(
    const unsigned short* __restrict__ Xb, const unsigned short* __restrict__ WtAll,
    const float* __restrict__ Bq, const float* __restrict__ Bk, const float* __restrict__ Bv,
    unsigned short* __restrict__ Qo, unsigned short* __restrict__ Ko,
    unsigned short* __restrict__ Vo)
{
    __shared__ unsigned short Xs[2][128 * 32];   // A tiles [row][k], ping-pong
    __shared__ unsigned short Ws[2][64 * 32];    // B tiles [n][k],  ping-pong

    const int tid  = threadIdx.x;
    const int wave = tid >> 5, lane = tid & 31;
    const int n0   = blockIdx.x * 64;
    const int m0   = blockIdx.y * 128;
    const int z    = blockIdx.z;
    const unsigned short* Wz = WtAll + (size_t)z * EMBED * EMBED;
    const float* Bias    = (z == 0) ? Bq : (z == 1) ? Bk : Bv;
    unsigned short* Out  = (z == 0) ? Qo : (z == 1) ? Ko : Vo;

    const int row16 = lane & 15;
    const int hiA = (lane >> 4) * 8, hiB = (lane >> 4) * 16;
    const int mrow = wave * 16;

    f32x8 zero = {0.f, 0.f, 0.f, 0.f, 0.f, 0.f, 0.f, 0.f};
    f32x8 acc[4];
#pragma unroll
    for (int t = 0; t < 4; ++t) acc[t] = zero;

#if HAVE_TDM
    // Prologue: start tile 0 transfers.
    if (wave == 0)      tdm_load_tile_b16(Xb + (size_t)m0 * EMBED, Xs[0], 32, 128, EMBED);
    else if (wave == 1) tdm_load_tile_b16(Wz + (size_t)n0 * EMBED, Ws[0], 32, 64, EMBED);
#endif

    for (int kk = 0; kk < EMBED; kk += 32) {
        const int ib = (kk >> 5) & 1;
        __syncthreads();   // everyone finished WMMAs on buf[1-ib]
#if HAVE_TDM
        const int nxt = kk + 32;
        if (nxt < EMBED) {
            if (wave == 0)
                tdm_load_tile_b16(Xb + (size_t)m0 * EMBED + nxt, Xs[1 - ib], 32, 128, EMBED);
            else if (wave == 1)
                tdm_load_tile_b16(Wz + (size_t)n0 * EMBED + nxt, Ws[1 - ib], 32, 64, EMBED);
        }
        if (wave < 2) {
            if (nxt < EMBED) tdm_wait_pipe();   // current tile landed, next in flight
            else             tdm_wait_all();
        }
#else
        copy_tile_b16(Xs[ib], Xb + (size_t)m0 * EMBED + kk, 128, 16, EMBED, tid);
        copy_tile_b16(Ws[ib], Wz + (size_t)n0 * EMBED + kk, 64, 16, EMBED, tid);
#endif
        __syncthreads();

        Frag16 a, bfr[4];
        load_a_frag(a, Xs[ib] + (mrow + row16) * 32, hiA);
#pragma unroll
        for (int t = 0; t < 4; ++t)
            load_b_frag(bfr[t], Ws[ib] + (t * 16 + row16) * 32 + hiB);
#pragma unroll
        for (int t = 0; t < 4; ++t) acc[t] = wmma_bf16(a, bfr[t], acc[t]);
    }

#pragma unroll
    for (int t = 0; t < 4; ++t) {
        int col = n0 + t * 16 + row16;
        float bias = Bias[col];
#pragma unroll
        for (int r = 0; r < 8; ++r) {
            int rowg = m0 + mrow + r + hiA;
            Out[(size_t)rowg * EMBED + col] = f2bf(acc[t][r] + bias);
        }
    }
}

// ---------------------------------------------------------------------------
// Kernel 2: flash attention. Block = (batch, 16-query tile), 8 waves; each
// wave owns a 128-wide d-slice. K tile streamed by double-buffered TDM; V tile
// transposed into LDS so PV B-fragments are contiguous. ~195KB dynamic LDS
// (CDNA5 allows up to 320KB per workgroup).
// ---------------------------------------------------------------------------
__global__ __launch_bounds__(256) void flash_attn_kernel(
    const unsigned short* __restrict__ Qw,
    const unsigned short* __restrict__ Kw,
    const unsigned short* __restrict__ Vw,
    unsigned short* __restrict__ Ao)
{
    extern __shared__ __align__(16) char smem[];
    unsigned short* Ks0 = (unsigned short*)smem;            // [32][1024] bf16
    unsigned short* Ks1 = Ks0 + 32 * 1024;                  // [32][1024] bf16
    unsigned short* Vt  = Ks1 + 32 * 1024;                  // [1024][32] bf16 (transposed)
    float*          Sc  = (float*)(Vt + 1024 * 32);         // [16][32] f32 scores
    unsigned short* Pp  = (unsigned short*)(Sc + 512);      // [16][32] bf16 probs
    float* rowm = (float*)(Pp + 512);
    float* rowl = rowm + 16;
    float* rowc = rowl + 16;

    const int tid  = threadIdx.x;
    const int wave = tid >> 5, lane = tid & 31;
    const int b    = blockIdx.y;
    const int q0   = blockIdx.x * 16;
    const int row16 = lane & 15;
    const int hiA = (lane >> 4) * 8, hiB = (lane >> 4) * 16;
    const int ds0 = wave * 128;
    const size_t baseQ = (size_t)(b * SEQ + q0) * EMBED;

    // Q fragments for this wave's d-slice stay in registers for all 64 steps.
    Frag16 qf[4];
#pragma unroll
    for (int kc = 0; kc < 4; ++kc)
        load_a_frag(qf[kc], Qw + baseQ + (size_t)row16 * EMBED + ds0 + kc * 32, hiA);

    f32x8 zero = {0.f, 0.f, 0.f, 0.f, 0.f, 0.f, 0.f, 0.f};
    f32x8 o[8];
#pragma unroll
    for (int t = 0; t < 8; ++t) o[t] = zero;

    if (tid < 16) { rowm[tid] = -1e30f; rowl[tid] = 0.f; }

#if HAVE_TDM
    // Prologue: start K tile 0 (one 1-D 64KB transfer).
    if (wave == 0)
        tdm_load_tile_b16(Kw + (size_t)(b * SEQ) * EMBED, Ks0, 32 * 1024, 1, 32 * 1024);
#endif

    for (int kt = 0; kt < SEQ / 32; ++kt) {
        const int kb = kt * 32;
        const int ib = kt & 1;
        unsigned short* Kcur = ib ? Ks1 : Ks0;
        __syncthreads();   // prior iter done with Ks[1-ib]/Vt/Pp/Sc

#if HAVE_TDM
        if (wave == 0 && kt + 1 < SEQ / 32)
            tdm_load_tile_b16(Kw + (size_t)(b * SEQ + kb + 32) * EMBED,
                              ib ? Ks0 : Ks1, 32 * 1024, 1, 32 * 1024);
#else
        copy_tile_b16(Kcur, Kw + (size_t)(b * SEQ + kb) * EMBED, 32, 512, EMBED, tid);
#endif
        // V tile transposed into Vt[d][key] so PV B-fragments are contiguous
        const unsigned int* Vsrc = (const unsigned int*)(Vw + (size_t)(b * SEQ + kb) * EMBED);
        for (int j = 0; j < 64; ++j) {
            int i = j * 256 + tid;
            int key = i >> 9, c2 = i & 511;
            unsigned int u = Vsrc[i];
            int d = c2 * 2;
            Vt[d * 32 + key]       = (unsigned short)(u & 0xffffu);
            Vt[(d + 1) * 32 + key] = (unsigned short)(u >> 16);
        }
        Sc[tid] = 0.f; Sc[tid + 256] = 0.f;
#if HAVE_TDM
        if (wave == 0) {
            if (kt + 1 < SEQ / 32) tdm_wait_pipe();   // tile kt landed, kt+1 in flight
            else                   tdm_wait_all();
        }
#endif
        __syncthreads();

        // Partial scores over this wave's 128-d slice, reduced via LDS atomics
#pragma unroll
        for (int ct = 0; ct < 2; ++ct) {
            Frag16 bf[4];
#pragma unroll
            for (int kc = 0; kc < 4; ++kc)
                load_b_frag(bf[kc], Kcur + (size_t)(ct * 16 + row16) * 1024 + ds0 + kc * 32 + hiB);
            f32x8 s = zero;
#pragma unroll
            for (int kc = 0; kc < 4; ++kc) s = wmma_bf16(qf[kc], bf[kc], s);
#pragma unroll
            for (int r = 0; r < 8; ++r)
                atomicAdd(&Sc[(r + hiA) * 32 + ct * 16 + row16], s[r]);
        }
        __syncthreads();

        // Online softmax: one thread per query row
        if (tid < 16) {
            const float scale = 0.03125f;   // 1/sqrt(1024)
            float m_old = rowm[tid];
            float srow[32];
            float mx = m_old;
#pragma unroll
            for (int c = 0; c < 32; ++c) {
                srow[c] = Sc[tid * 32 + c] * scale;
                mx = fmaxf(mx, srow[c]);
            }
            float corr = __expf(m_old - mx);
            float sum = 0.f;
#pragma unroll
            for (int c = 0; c < 32; ++c) {
                float p = __expf(srow[c] - mx);
                sum += p;
                Pp[tid * 32 + c] = f2bf(p);
            }
            rowm[tid] = mx;
            rowl[tid] = rowl[tid] * corr + sum;
            rowc[tid] = corr;
        }
        __syncthreads();

        // Rescale accumulators then O += P * V
        float cf[8];
#pragma unroll
        for (int r = 0; r < 8; ++r) cf[r] = rowc[r + hiA];
#pragma unroll
        for (int t = 0; t < 8; ++t)
#pragma unroll
            for (int r = 0; r < 8; ++r) o[t][r] *= cf[r];

        Frag16 pf;
        load_a_frag(pf, Pp + row16 * 32, hiA);
#pragma unroll
        for (int t = 0; t < 8; t += 2) {
            Frag16 v0, v1;
            load_b_frag(v0, Vt + (size_t)(ds0 + t * 16 + row16) * 32 + hiB);
            load_b_frag(v1, Vt + (size_t)(ds0 + (t + 1) * 16 + row16) * 32 + hiB);
            o[t]     = wmma_bf16(pf, v0, o[t]);
            o[t + 1] = wmma_bf16(pf, v1, o[t + 1]);
        }
    }
    __syncthreads();

    float lin[8];
#pragma unroll
    for (int r = 0; r < 8; ++r) lin[r] = 1.f / rowl[r + hiA];
#pragma unroll
    for (int t = 0; t < 8; ++t)
#pragma unroll
        for (int r = 0; r < 8; ++r) {
            int row = r + hiA;
            Ao[baseQ + (size_t)row * EMBED + ds0 + t * 16 + row16] = f2bf(o[t][r] * lin[r]);
        }
}

// ---------------------------------------------------------------------------
// Kernel 3: out = attn*Wo + bo + X residual (f32 into d_out). grid (16, 64).
// Same double-buffered TDM pipeline as kernel 1.
// ---------------------------------------------------------------------------
__global__ __launch_bounds__(256) void proj_gemm_kernel(
    const unsigned short* __restrict__ Aw, const unsigned short* __restrict__ Wto,
    const float* __restrict__ Bo, const float* __restrict__ X, float* __restrict__ Out)
{
    __shared__ unsigned short As[2][128 * 32];
    __shared__ unsigned short Ws[2][64 * 32];

    const int tid  = threadIdx.x;
    const int wave = tid >> 5, lane = tid & 31;
    const int n0   = blockIdx.x * 64;
    const int m0   = blockIdx.y * 128;
    const int row16 = lane & 15;
    const int hiA = (lane >> 4) * 8, hiB = (lane >> 4) * 16;
    const int mrow = wave * 16;

    f32x8 zero = {0.f, 0.f, 0.f, 0.f, 0.f, 0.f, 0.f, 0.f};
    f32x8 acc[4];
#pragma unroll
    for (int t = 0; t < 4; ++t) acc[t] = zero;

#if HAVE_TDM
    if (wave == 0)      tdm_load_tile_b16(Aw + (size_t)m0 * EMBED, As[0], 32, 128, EMBED);
    else if (wave == 1) tdm_load_tile_b16(Wto + (size_t)n0 * EMBED, Ws[0], 32, 64, EMBED);
#endif

    for (int kk = 0; kk < EMBED; kk += 32) {
        const int ib = (kk >> 5) & 1;
        __syncthreads();
#if HAVE_TDM
        const int nxt = kk + 32;
        if (nxt < EMBED) {
            if (wave == 0)
                tdm_load_tile_b16(Aw + (size_t)m0 * EMBED + nxt, As[1 - ib], 32, 128, EMBED);
            else if (wave == 1)
                tdm_load_tile_b16(Wto + (size_t)n0 * EMBED + nxt, Ws[1 - ib], 32, 64, EMBED);
        }
        if (wave < 2) {
            if (nxt < EMBED) tdm_wait_pipe();
            else             tdm_wait_all();
        }
#else
        copy_tile_b16(As[ib], Aw + (size_t)m0 * EMBED + kk, 128, 16, EMBED, tid);
        copy_tile_b16(Ws[ib], Wto + (size_t)n0 * EMBED + kk, 64, 16, EMBED, tid);
#endif
        __syncthreads();

        Frag16 a, bfr[4];
        load_a_frag(a, As[ib] + (mrow + row16) * 32, hiA);
#pragma unroll
        for (int t = 0; t < 4; ++t)
            load_b_frag(bfr[t], Ws[ib] + (t * 16 + row16) * 32 + hiB);
#pragma unroll
        for (int t = 0; t < 4; ++t) acc[t] = wmma_bf16(a, bfr[t], acc[t]);
    }

#pragma unroll
    for (int t = 0; t < 4; ++t) {
        int col = n0 + t * 16 + row16;
        float bias = Bo[col];
#pragma unroll
        for (int r = 0; r < 8; ++r) {
            size_t idx = (size_t)(m0 + mrow + r + hiA) * EMBED + col;
            Out[idx] = acc[t][r] + bias + X[idx];   // residual fused
        }
    }
}

// ---------------------------------------------------------------------------
// Kernel 4: in-place LayerNorm over each row of d_out. Block = one row.
// ---------------------------------------------------------------------------
__global__ __launch_bounds__(256) void layernorm_kernel(
    float* __restrict__ Y, const float* __restrict__ gamma,
    const float* __restrict__ beta)
{
    __shared__ float sred[256], sred2[256];
    __shared__ float smu, srs;
    const int row = blockIdx.x, tid = threadIdx.x;
    float v[4], s = 0.f, s2 = 0.f;
#pragma unroll
    for (int j = 0; j < 4; ++j) {
        v[j] = Y[(size_t)row * EMBED + j * 256 + tid];
        s += v[j]; s2 += v[j] * v[j];
    }
    sred[tid] = s; sred2[tid] = s2;
    __syncthreads();
    for (int off = 128; off > 0; off >>= 1) {
        if (tid < off) { sred[tid] += sred[tid + off]; sred2[tid] += sred2[tid + off]; }
        __syncthreads();
    }
    if (tid == 0) {
        float mean = sred[0] * (1.f / EMBED);
        float var  = sred2[0] * (1.f / EMBED) - mean * mean;   // population variance
        smu = mean; srs = rsqrtf(var + 1e-6f);
    }
    __syncthreads();
    float mean = smu, rs = srs;
#pragma unroll
    for (int j = 0; j < 4; ++j) {
        int col = j * 256 + tid;
        Y[(size_t)row * EMBED + col] = (v[j] - mean) * rs * gamma[col] + beta[col];
    }
}

// ---------------------------------------------------------------------------
extern "C" void kernel_launch(void* const* d_in, const int* in_sizes, int n_in,
                              void* d_out, int out_size, void* d_ws, size_t ws_size,
                              hipStream_t stream) {
    (void)in_sizes; (void)n_in; (void)out_size; (void)ws_size;
    const float* X     = (const float*)d_in[0];
    const float* wq    = (const float*)d_in[1];
    const float* bq    = (const float*)d_in[2];
    const float* wk    = (const float*)d_in[3];
    const float* bk    = (const float*)d_in[4];
    const float* wv    = (const float*)d_in[5];
    const float* bv    = (const float*)d_in[6];
    const float* wo    = (const float*)d_in[7];
    const float* bo    = (const float*)d_in[8];
    const float* gamma = (const float*)d_in[9];
    const float* beta  = (const float*)d_in[10];
    float* Out = (float*)d_out;

    // Workspace (bf16): Q,K,V,attn,Xbf = 5*16MB; Wt (4 transposed weights) 8MB.
    const size_t NE = (size_t)MTOT * EMBED;
    unsigned short* Q  = (unsigned short*)d_ws;
    unsigned short* K  = Q  + NE;
    unsigned short* V  = K  + NE;
    unsigned short* A  = V  + NE;
    unsigned short* Xb = A  + NE;
    unsigned short* Wt = Xb + NE;   // 4 * 1024 * 1024 elements

    cvt_bf16_kernel<<<dim3((int)(NE / 2 / 256)), 256, 0, stream>>>(
        (const float2*)X, (unsigned int*)Xb, (int)(NE / 2));
    wtrans_kernel<<<dim3(16, 16, 4), 256, 0, stream>>>(wq, wk, wv, wo, Wt);

    qkv_gemm_kernel<<<dim3(EMBED / 64, MTOT / 128, 3), 256, 0, stream>>>(
        Xb, Wt, bq, bk, bv, Q, K, V);

    const size_t attn_lds = (size_t)(2 * 32 * 1024 + 1024 * 32) * 2 + 512 * 4 + 512 * 2 + 64 * 4;
    (void)hipFuncSetAttribute(reinterpret_cast<const void*>(flash_attn_kernel),
                              hipFuncAttributeMaxDynamicSharedMemorySize, (int)attn_lds);
    flash_attn_kernel<<<dim3(SEQ / 16, NB), 256, attn_lds, stream>>>(Q, K, V, A);

    proj_gemm_kernel<<<dim3(EMBED / 64, MTOT / 128), 256, 0, stream>>>(
        A, Wt + (size_t)3 * EMBED * EMBED, bo, X, Out);

    layernorm_kernel<<<dim3(MTOT), 256, 0, stream>>>(Out, gamma, beta);
}
